// SQN_3075196584361
// MI455X (gfx1250) — matrix operand order repaired and checked by reading, not compiled, not run on hardware
//
#include <hip/hip_runtime.h>
#include <hip/hip_bf16.h>

// ---------------------------------------------------------------------------
// CDNA5 / gfx1250 SNN forward (LIF, T=100), wave32.
//  * h0 = input @ W0 hoisted out of the T-loop (exact f32, once).
//  * Binary spikes -> exact in f16 -> recurrent GEMMs on v_wmma_f32_16x16x32_f16.
//  * B tiles staged in LDS, double-buffered with GLOBAL_LOAD_ASYNC_TO_LDS_B128
//    (ASYNCcnt) when available; ds-store fallback otherwise.
//  * syn updates fused into GEMM epilogues; both per-step GEMMs in one launch.
// ---------------------------------------------------------------------------

typedef __attribute__((ext_vector_type(16))) _Float16 v16h;
typedef __attribute__((ext_vector_type(8)))  float    v8f;

#define ALPHA 0.90f
#define BETA  0.85f
#define T_STEPS 100
#define Bsz  1024      // batch
#define N0   784       // input width
#define N1   1024      // hidden width
#define N2P  16        // output width padded (true = 10)

union FragH {           // one lane's 16 f16 elements of a WMMA fragment
    v16h  v;
    uint4 q[2];
};

// ---- async Global->LDS copy (ASYNCcnt) with safe fallback ------------------
#if defined(__has_builtin)
#  if __has_builtin(__builtin_amdgcn_global_load_async_to_lds_b128) && \
      __has_builtin(__builtin_amdgcn_s_wait_asynccnt)
#    define HAVE_ASYNC_LDS 1
#  endif
#endif

#ifdef HAVE_ASYNC_LDS
// Builtin signature (from hipcc diagnostic): first param is a pointer to a
// 16-byte int vector in the global (__device__/AS1) address space; second is
// the LDS (__shared__/AS3) destination; then imm offset and imm cpol.
typedef int vint4 __attribute__((vector_size(16)));
typedef __attribute__((address_space(1))) vint4 gvint4;
typedef __attribute__((address_space(3))) vint4 lvint4;
#  define ASYNC_COPY16(dst_lds, src_glb)                                      \
     __builtin_amdgcn_global_load_async_to_lds_b128(                          \
         (gvint4*)(src_glb), (lvint4*)(dst_lds), 0, 0)
#  define ASYNC_WAIT() __builtin_amdgcn_s_wait_asynccnt(0)
#else
#  define ASYNC_COPY16(dst_lds, src_glb)                                      \
     (*(uint4*)(dst_lds) = *(const uint4*)(src_glb))
#  define ASYNC_WAIT() ((void)0)
#endif

#define WMMA_F16(ACC, A, B) \
    ACC = __builtin_amdgcn_wmma_f32_16x16x32_f16(false, (A), false, (B), (short)0, (ACC), false, false)

// ---------------------------------------------------------------------------
// Init: zero state, convert W1 -> f16, convert+pad W2 -> f16 (1024 x 16).
// grid 4096 x 256 == exactly Bsz*N1 threads.
// ---------------------------------------------------------------------------
__global__ void init_state(const float* __restrict__ W1, const float* __restrict__ W2,
                           float* __restrict__ syn0, float* __restrict__ mem0,
                           float* __restrict__ syn1, float* __restrict__ mem1,
                           float* __restrict__ syn2, float* __restrict__ mem2,
                           _Float16* __restrict__ W1h, _Float16* __restrict__ W2h)
{
    int i = blockIdx.x * blockDim.x + threadIdx.x;        // 0 .. 1M-1
    syn0[i] = 0.0f; mem0[i] = 0.0f;
    syn1[i] = 0.0f; mem1[i] = 0.0f;
    W1h[i] = (_Float16)W1[i];
    if (i < Bsz * N2P) {
        syn2[i] = 0.0f; mem2[i] = 0.0f;
        int k = i >> 4, n = i & 15;                        // W2h[k][n], n padded
        W2h[i] = (n < 10) ? (_Float16)W2[k * 10 + n] : (_Float16)0.0f;
    }
}

// ---------------------------------------------------------------------------
// One-time exact-f32 GEMM: H0[1024x1024] = X[1024x784] @ W0[784x1024].
// ---------------------------------------------------------------------------
__global__ void gemm_h0(const float* __restrict__ X, const float* __restrict__ W0,
                        float* __restrict__ H0)
{
    __shared__ float As[16][16];
    __shared__ float Bs[16][17];
    const int tx = threadIdx.x, ty = threadIdx.y;
    const int row = blockIdx.y * 16 + ty;
    const int col = blockIdx.x * 16 + tx;
    float s = 0.0f;
    for (int k0 = 0; k0 < N0; k0 += 16) {                  // 784 = 49 * 16
        As[ty][tx] = X[row * N0 + k0 + tx];
        Bs[ty][tx] = W0[(k0 + ty) * N1 + col];
        __syncthreads();
#pragma unroll
        for (int kk = 0; kk < 16; ++kk) s += As[ty][kk] * Bs[kk][tx];
        __syncthreads();
    }
    H0[row * N1 + col] = s;
}

// ---------------------------------------------------------------------------
// Per-step elementwise LIF update (spikes from OLD membranes, membranes from
// OLD syn). syn1/syn2 are updated afterwards by the fused-GEMM kernel.
// ---------------------------------------------------------------------------
__global__ void lif_update(const float* __restrict__ H0,
                           float* __restrict__ syn0, float* __restrict__ mem0,
                           const float* __restrict__ syn1, float* __restrict__ mem1,
                           const float* __restrict__ syn2, float* __restrict__ mem2,
                           _Float16* __restrict__ S0h, _Float16* __restrict__ S1h)
{
    int i = blockIdx.x * blockDim.x + threadIdx.x;
    float m0 = mem0[i], s0 = syn0[i];
    float sp0 = ((m0 - 1.0f) > 0.0f) ? 1.0f : 0.0f;
    S0h[i] = (_Float16)sp0;
    mem0[i] = BETA * m0 + s0 - sp0;
    syn0[i] = ALPHA * s0 + H0[i];          // h0 is timestep-invariant
    float m1 = mem1[i];
    float sp1 = ((m1 - 1.0f) > 0.0f) ? 1.0f : 0.0f;
    S1h[i] = (_Float16)sp1;
    mem1[i] = BETA * m1 + syn1[i] - sp1;
    if (i < Bsz * N2P) mem2[i] = BETA * mem2[i] + syn2[i];
}

// ---------------------------------------------------------------------------
// Fused per-step GEMMs (one launch, 136 blocks x 256 threads):
//   blocks [0,128):  syn1[1024x1024] = ALPHA*syn1 + S0h @ W1h
//                    128(M) x 64(N) tile, 8 waves x (16x64); B tile staged in
//                    LDS via double-buffered async Global->LDS copies.
//   blocks [128,136): syn2[1024x16]  = ALPHA*syn2 + S1h @ W2h (tiny, direct).
// ---------------------------------------------------------------------------
__global__ void __launch_bounds__(256)
step_gemms(const _Float16* __restrict__ S0h, const _Float16* __restrict__ W1h,
           float* __restrict__ syn1,
           const _Float16* __restrict__ S1h, const _Float16* __restrict__ W2h,
           float* __restrict__ syn2)
{
    // 2 buffers x 32(K) x 72(N, padded: stride 144B -> conflict-free rows)
    __shared__ __align__(16) _Float16 Bs[2][32][72];

    const int lane = threadIdx.x & 31;
    const int wave = threadIdx.x >> 5;
    const int g = lane >> 4, r = lane & 15;

    if (blockIdx.x < 128) {
        // ----- main spike GEMM -----
        const int n0 = (blockIdx.x & 15) * 64;
        const int m0 = (blockIdx.x >> 4) * 128 + wave * 16;

        // cooperative B-tile staging: 256 threads x 16B = 32x64 f16 tile
        const int krow = threadIdx.x >> 3;          // 0..31
        const int col8 = (threadIdx.x & 7) << 3;    // 0,8,..,56
        auto stage = [&](int buf, int k0) {
            const _Float16* src = W1h + (size_t)(k0 + krow) * N1 + n0 + col8;
            ASYNC_COPY16(&Bs[buf][krow][col8], src);
        };

        v8f acc0 = {}, acc1 = {}, acc2 = {}, acc3 = {};
        const _Float16* Arow = S0h + (size_t)(m0 + r) * N1;
        // warm the output tile (read-modify-write in the epilogue)
        __builtin_prefetch(syn1 + (size_t)(m0 + r) * N1 + n0, 1, 3);

        stage(0, 0);
        int it = 0;
        for (int k0 = 0; k0 < N1; k0 += 32, it ^= 1) {
            ASYNC_WAIT();               // this wave's copies for buffer `it` done
            __syncthreads();            // ... and everyone else's
            if (k0 + 32 < N1) stage(it ^ 1, k0 + 32);   // overlap next copy

            FragH a, b0, b1, b2, b3;
            a.q[0] = *(const uint4*)(Arow + k0 + g * 8);
            a.q[1] = *(const uint4*)(Arow + k0 + 16 + g * 8);
#pragma unroll
            for (int s = 0; s < 16; ++s) {
                const int kk = (s < 8) ? (g * 8 + s) : (8 + g * 8 + s);
                b0.v[s] = Bs[it][kk][r];
                b1.v[s] = Bs[it][kk][16 + r];
                b2.v[s] = Bs[it][kk][32 + r];
                b3.v[s] = Bs[it][kk][48 + r];
            }
            WMMA_F16(acc0, a.v, b0.v);
            WMMA_F16(acc1, a.v, b1.v);
            WMMA_F16(acc2, a.v, b2.v);
            WMMA_F16(acc3, a.v, b3.v);
            __syncthreads();            // all waves done reading Bs[it]
        }
        // D layout: VGPR i -> row m0 + 8*g + i, col n0 + r (+16 per sub-tile)
#pragma unroll
        for (int i = 0; i < 8; ++i) {
            float* Cr = syn1 + (size_t)(m0 + g * 8 + i) * N1 + n0 + r;
            Cr[0]  = ALPHA * Cr[0]  + acc0[i];
            Cr[16] = ALPHA * Cr[16] + acc1[i];
            Cr[32] = ALPHA * Cr[32] + acc2[i];
            Cr[48] = ALPHA * Cr[48] + acc3[i];
        }
    } else {
        // ----- output-layer GEMM (N padded to 16) -----
        const int m0 = (blockIdx.x - 128) * 128 + wave * 16;
        v8f acc = {};
        const _Float16* Arow = S1h + (size_t)(m0 + r) * N1;
        for (int k = 0; k < N1; k += 32) {
            FragH a, b;
            a.q[0] = *(const uint4*)(Arow + k + g * 8);
            a.q[1] = *(const uint4*)(Arow + k + 16 + g * 8);
            const _Float16* Bcol = W2h + (size_t)k * N2P + r;
#pragma unroll
            for (int s = 0; s < 16; ++s) {
                const int kk = (s < 8) ? (g * 8 + s) : (8 + g * 8 + s);
                b.v[s] = Bcol[kk * N2P];
            }
            WMMA_F16(acc, a.v, b.v);
        }
#pragma unroll
        for (int i = 0; i < 8; ++i) {
            float* Cr = syn2 + (size_t)(m0 + g * 8 + i) * N2P + r;
            *Cr = ALPHA * (*Cr) + acc[i];
        }
    }
}

// Strip the N=16 padding: out[b][0..9] = mem2[b][0..9]
__global__ void emit_output(const float* __restrict__ mem2, float* __restrict__ out)
{
    int i = blockIdx.x * blockDim.x + threadIdx.x;
    if (i < Bsz * 10) {
        int b = i / 10, j = i % 10;
        out[i] = mem2[b * N2P + j];
    }
}

// ---------------------------------------------------------------------------
extern "C" void kernel_launch(void* const* d_in, const int* in_sizes, int n_in,
                              void* d_out, int out_size, void* d_ws, size_t ws_size,
                              hipStream_t stream)
{
    (void)in_sizes; (void)n_in; (void)out_size; (void)ws_size;
    const float* X  = (const float*)d_in[0];   // 1024 x 784
    const float* W0 = (const float*)d_in[1];   // 784  x 1024
    const float* W1 = (const float*)d_in[2];   // 1024 x 1024
    const float* W2 = (const float*)d_in[3];   // 1024 x 10
    float* out = (float*)d_out;                // 1024 x 10

    char* ws = (char*)d_ws;
    const size_t MB = 1u << 20, KB = 1u << 10;
    float*    H0   = (float*)(ws);                          // 4 MB
    float*    syn0 = (float*)(ws +  4 * MB);                // 4 MB
    float*    mem0 = (float*)(ws +  8 * MB);                // 4 MB
    float*    syn1 = (float*)(ws + 12 * MB);                // 4 MB
    float*    mem1 = (float*)(ws + 16 * MB);                // 4 MB
    float*    syn2 = (float*)(ws + 20 * MB);                // 64 KB (padded N=16)
    float*    mem2 = (float*)(ws + 20 * MB +  64 * KB);     // 64 KB
    _Float16* S0h  = (_Float16*)(ws + 20 * MB + 128 * KB);  // 2 MB
    _Float16* S1h  = (_Float16*)(ws + 22 * MB + 128 * KB);  // 2 MB
    _Float16* W1h  = (_Float16*)(ws + 24 * MB + 128 * KB);  // 2 MB
    _Float16* W2h  = (_Float16*)(ws + 26 * MB + 128 * KB);  // 32 KB

    // 1) zero state + weight conversion (re-done every call: deterministic)
    init_state<<<4096, 256, 0, stream>>>(W1, W2, syn0, mem0, syn1, mem1,
                                         syn2, mem2, W1h, W2h);
    // 2) hoisted layer-0 GEMM (exact f32, once)
    gemm_h0<<<dim3(64, 64), dim3(16, 16), 0, stream>>>(X, W0, H0);

    // 3) recurrence: 100 steps x (elementwise LIF, fused WMMA GEMMs)
    for (int t = 0; t < T_STEPS; ++t) {
        lif_update<<<4096, 256, 0, stream>>>(H0, syn0, mem0, syn1, mem1,
                                             syn2, mem2, S0h, S1h);
        step_gemms<<<136, 256, 0, stream>>>(S0h, W1h, syn1, S1h, W2h, syn2);
    }
    // 4) final membrane potential of the output layer (strip padding)
    emit_output<<<(Bsz * 10 + 255) / 256, 256, 0, stream>>>(mem2, out);
}
